// Attention_73151882985581
// MI455X (gfx1250) — matrix operand order repaired and checked by reading, not compiled, not run on hardware
//
#include <hip/hip_runtime.h>
#include <hip/hip_bf16.h>

typedef __attribute__((ext_vector_type(16))) _Float16 v16h;
typedef __attribute__((ext_vector_type(8)))  _Float16 v8h;
typedef __attribute__((ext_vector_type(4)))  _Float16 v4h;
typedef __attribute__((ext_vector_type(8)))  float    v8f;
typedef __attribute__((ext_vector_type(4)))  float    v4f;
typedef int v4i_b128 __attribute__((vector_size(16)));   // b128 payload type for async LDS builtin

#define BQ 2
#define SEQ 2048
#define DMODEL 2048
#define NH 16
#define NKV 4
#define HD 128
#define KVDIM (NKV*HD)      // 512
#define MTOT (BQ*SEQ)       // 4096

// ---- CDNA5 async global->LDS path (guarded; falls back to sync staging) ----
#if defined(__gfx1250__) && __has_builtin(__builtin_amdgcn_global_load_async_to_lds_b128)
#define HAVE_ASYNC_LDS 1
#else
#define HAVE_ASYNC_LDS 0
#endif

#if HAVE_ASYNC_LDS
#define ASYNC_COPY_B128(gsrc, ldst)                                          \
  __builtin_amdgcn_global_load_async_to_lds_b128(                            \
      (__attribute__((address_space(1))) v4i_b128*)(gsrc),                   \
      (__attribute__((address_space(3))) v4i_b128*)(ldst), 0, 0)
#if __has_builtin(__builtin_amdgcn_s_wait_asynccnt)
#define ASYNC_WAIT() __builtin_amdgcn_s_wait_asynccnt(0)
#else
#define ASYNC_WAIT() asm volatile("s_wait_asynccnt 0x0" ::: "memory")
#endif
#else
#define ASYNC_WAIT()
#endif

__device__ __forceinline__ v8f wmma_f16(v16h a, v16h b, v8f c) {
  // D = A(16x32 f16) * B(32x16 f16) + C(16x16 f32)
  return __builtin_amdgcn_wmma_f32_16x16x32_f16(
      /*neg_a=*/false, a, /*neg_b=*/false, b,
      /*c_mod=*/(short)0, c, /*reuse_a=*/false, /*reuse_b=*/false);
}

// A-operand fragment: lane holds row M=l16; K runs [lh*16, lh*16+16)
__device__ __forceinline__ v16h load_a_frag(const _Float16* rowbase, int lh) {
  union { v16h v; v8h h8[2]; } u;
  u.h8[0] = *(const v8h*)(rowbase + lh * 16);
  u.h8[1] = *(const v8h*)(rowbase + lh * 16 + 8);
  return u.v;
}

// ---------------------------------------------------------------- convert f32 -> f16
__global__ void cvt_f32_to_f16(const float* __restrict__ s, _Float16* __restrict__ d, size_t n) {
  size_t i = ((size_t)blockIdx.x * blockDim.x + threadIdx.x) * 4;
  if (i + 3 < n) {
    v4f f = *(const v4f*)(s + i);
    v4h h;
#pragma unroll
    for (int j = 0; j < 4; ++j) h[j] = (_Float16)f[j];
    *(v4h*)(d + i) = h;
  }
}

// ---------------------------------------------------------------- RoPE (in place on f16)
__global__ void rope_kernel(_Float16* __restrict__ x, const float* __restrict__ fcos,
                            const float* __restrict__ fsin, int n_heads) {
  const int stride = n_heads * HD;
  size_t idx = (size_t)blockIdx.x * blockDim.x + threadIdx.x;
  size_t total = (size_t)MTOT * n_heads * (HD / 2);
  if (idx >= total) return;
  int i    = (int)(idx % (HD / 2));
  int head = (int)((idx / (HD / 2)) % n_heads);
  size_t row = idx / ((size_t)(HD / 2) * n_heads);
  int s = (int)(row % SEQ);
  float c  = fcos[s * (HD / 2) + i];
  float sn = fsin[s * (HD / 2) + i];
  _Float16* p = x + row * stride + head * HD + 2 * i;
  float xr = (float)p[0], xi = (float)p[1];
  p[0] = (_Float16)(xr * c - xi * sn);
  p[1] = (_Float16)(xr * sn + xi * c);
}

// ---------------------------------------------------------------- tiled WMMA GEMM
// C[M,N] = A[M,K](f16) * B[K,N](f16); 128x128 block tile, BK=64, 8 waves (4x2), 32x64/wave
template <bool OUT_F32>
__global__ __launch_bounds__(256) void gemm_f16_kernel(const _Float16* __restrict__ A,
                                                       const _Float16* __restrict__ B,
                                                       void* __restrict__ out,
                                                       int K, int N) {
  __shared__ __align__(128) _Float16 As[128][64];
  __shared__ __align__(128) _Float16 BsT[128][64];   // [n][k]
  const int tid  = threadIdx.x;
  const int wave = tid >> 5, lane = tid & 31;
  const int lh   = lane >> 4, l16 = lane & 15;
  const int wm   = wave & 3, wn = wave >> 2;
  const int rowBase = blockIdx.y * 128;
  const int colBase = blockIdx.x * 128;

  v8f acc[2][4];
#pragma unroll
  for (int i = 0; i < 2; ++i)
#pragma unroll
    for (int j = 0; j < 4; ++j)
#pragma unroll
      for (int e = 0; e < 8; ++e) acc[i][j][e] = 0.0f;

  for (int kb = 0; kb < K; kb += 64) {
    // stage A tile: 128x64 halves = 1024 chunks of 8 halves, 4 per thread
#pragma unroll
    for (int c = 0; c < 4; ++c) {
      int idx = tid + c * 256;
      int r = idx >> 3, col = (idx & 7) * 8;
      const _Float16* gp = A + (size_t)(rowBase + r) * K + kb + col;
#if HAVE_ASYNC_LDS
      ASYNC_COPY_B128(gp, &As[r][col]);
#else
      *(v8h*)&As[r][col] = *(const v8h*)gp;
#endif
      __builtin_prefetch(gp + 64, 0, 0);     // next K-slab -> global_prefetch_b8
    }
    // stage B tile transposed: 64x128 -> BsT[n][k], 1024 chunks, 4 per thread
#pragma unroll
    for (int c = 0; c < 4; ++c) {
      int idx = tid + c * 256;
      int kr = idx >> 4, n0 = (idx & 15) * 8;
      const _Float16* gp = B + (size_t)(kb + kr) * N + colBase + n0;
      v8h bv = *(const v8h*)gp;
      __builtin_prefetch(gp + (size_t)64 * N, 0, 0);
#pragma unroll
      for (int i = 0; i < 8; ++i) BsT[n0 + i][kr] = bv[i];
    }
    ASYNC_WAIT();
    __syncthreads();

#pragma unroll
    for (int kk = 0; kk < 2; ++kk) {
      v16h af[2], bf[4];
#pragma unroll
      for (int i = 0; i < 2; ++i)
        af[i] = load_a_frag(&As[wm * 32 + i * 16 + l16][kk * 32], lh);
#pragma unroll
      for (int j = 0; j < 4; ++j)
        bf[j] = *(const v16h*)&BsT[wn * 64 + j * 16 + l16][kk * 32 + lh * 16];
#pragma unroll
      for (int i = 0; i < 2; ++i)
#pragma unroll
        for (int j = 0; j < 4; ++j)
          acc[i][j] = wmma_f16(af[i], bf[j], acc[i][j]);
    }
    __syncthreads();
  }

#pragma unroll
  for (int i = 0; i < 2; ++i)
#pragma unroll
    for (int j = 0; j < 4; ++j)
#pragma unroll
      for (int e = 0; e < 8; ++e) {
        int r  = rowBase + wm * 32 + i * 16 + e + lh * 8;
        int cN = colBase + wn * 64 + j * 16 + l16;
        if (OUT_F32) ((float*)out)[(size_t)r * N + cN] = acc[i][j][e];
        else         ((_Float16*)out)[(size_t)r * N + cN] = (_Float16)acc[i][j][e];
      }
}

// ---------------------------------------------------------------- causal flash attention
// grid: (S/128, NH, B); 8 waves, wave w owns 16 query rows; key tiles of 64 staged in LDS
__global__ __launch_bounds__(256) void flash_kernel(const _Float16* __restrict__ q,
                                                    const _Float16* __restrict__ k,
                                                    const _Float16* __restrict__ v,
                                                    _Float16* __restrict__ ao) {
  __shared__ __align__(128) _Float16 Ks[64][HD];      // [key][d]
  __shared__ __align__(128) _Float16 Vt[HD][64];      // [d][key]
  __shared__ __align__(128) _Float16 Ps[8][16][64];   // per-wave P tile
  const int qb = blockIdx.x, h = blockIdx.y, b = blockIdx.z;
  const int g  = h >> 2;                               // kv head = h / N_REP
  const int tid = threadIdx.x, wave = tid >> 5, lane = tid & 31;
  const int lh = lane >> 4, l16 = lane & 15;
  const int qr0 = qb * 128 + wave * 16;
  const float scale = 0.08838834764831845f;            // 1/sqrt(128)

  v16h qf[4];
  {
    const _Float16* qp = q + (size_t)(b * SEQ + qr0 + l16) * DMODEL + h * HD;
#pragma unroll
    for (int t = 0; t < 4; ++t) qf[t] = load_a_frag(qp + t * 32, lh);
  }
  v8f o[8];
  float m[8], l[8];
#pragma unroll
  for (int t = 0; t < 8; ++t) {
#pragma unroll
    for (int e = 0; e < 8; ++e) o[t][e] = 0.0f;
    m[t] = -3.0e38f;
    l[t] = 0.0f;
  }

  const int kend = qb * 128 + 128;
  for (int k0 = 0; k0 < kend; k0 += 64) {
    // stage K (async, row major) and V (transposed): 64 rows x 128 d
#pragma unroll
    for (int c = 0; c < 4; ++c) {
      int idx = tid + c * 256;
      int r = idx >> 4, d0 = (idx & 15) * 8;
      size_t base = (size_t)(b * SEQ + k0 + r) * KVDIM + g * HD + d0;
#if HAVE_ASYNC_LDS
      ASYNC_COPY_B128(k + base, &Ks[r][d0]);
#else
      *(v8h*)&Ks[r][d0] = *(const v8h*)(k + base);
#endif
      v8h vv = *(const v8h*)(v + base);
      __builtin_prefetch(k + base + (size_t)64 * KVDIM, 0, 0);
      __builtin_prefetch(v + base + (size_t)64 * KVDIM, 0, 0);
#pragma unroll
      for (int i = 0; i < 8; ++i) Vt[d0 + i][r] = vv[i];
    }
    ASYNC_WAIT();
    __syncthreads();

    if (k0 <= qr0 + 15) {                       // wave-uniform: tile not fully masked
      v8f sa[4];
#pragma unroll
      for (int s = 0; s < 4; ++s)
#pragma unroll
        for (int e = 0; e < 8; ++e) sa[s][e] = 0.0f;
      // scores: S = Q * K^T ; B[k=d][n=key] = Ks[key][d]
#pragma unroll
      for (int sub = 0; sub < 4; ++sub)
#pragma unroll
        for (int t = 0; t < 4; ++t) {
          v16h bf = *(const v16h*)&Ks[sub * 16 + l16][t * 32 + lh * 16];
          sa[sub] = wmma_f16(qf[t], bf, sa[sub]);
        }
      // scale + causal mask + row max
      float rm[8];
#pragma unroll
      for (int e = 0; e < 8; ++e) rm[e] = -3.0e38f;
#pragma unroll
      for (int sub = 0; sub < 4; ++sub)
#pragma unroll
        for (int e = 0; e < 8; ++e) {
          float val = sa[sub][e] * scale;
          int qrow = qr0 + e + lh * 8;
          int key  = k0 + sub * 16 + l16;
          if (key > qrow) val = -3.0e38f;
          sa[sub][e] = val;
          rm[e] = fmaxf(rm[e], val);
        }
#pragma unroll
      for (int e = 0; e < 8; ++e) {
        rm[e] = fmaxf(rm[e], __shfl_xor(rm[e], 1, 32));
        rm[e] = fmaxf(rm[e], __shfl_xor(rm[e], 2, 32));
        rm[e] = fmaxf(rm[e], __shfl_xor(rm[e], 4, 32));
        rm[e] = fmaxf(rm[e], __shfl_xor(rm[e], 8, 32));
      }
      float fac[8], rs[8];
#pragma unroll
      for (int e = 0; e < 8; ++e) {
        float mn = fmaxf(m[e], rm[e]);
        fac[e] = __expf(m[e] - mn);
        m[e] = mn;
        rs[e] = 0.0f;
      }
      // P = exp(S - m); park in per-wave LDS to re-fragment as A operand
#pragma unroll
      for (int sub = 0; sub < 4; ++sub)
#pragma unroll
        for (int e = 0; e < 8; ++e) {
          float p = __expf(sa[sub][e] - m[e]);
          rs[e] += p;
          Ps[wave][e + lh * 8][sub * 16 + l16] = (_Float16)p;
        }
#pragma unroll
      for (int e = 0; e < 8; ++e) {
        rs[e] += __shfl_xor(rs[e], 1, 32);
        rs[e] += __shfl_xor(rs[e], 2, 32);
        rs[e] += __shfl_xor(rs[e], 4, 32);
        rs[e] += __shfl_xor(rs[e], 8, 32);
        l[e] = l[e] * fac[e] + rs[e];
      }
#pragma unroll
      for (int t = 0; t < 8; ++t)
#pragma unroll
        for (int e = 0; e < 8; ++e) o[t][e] *= fac[e];
      // wave-local LDS store->load fence (gfx1250 split DS counter)
      asm volatile("s_wait_dscnt 0" ::: "memory");
      v16h pa[2];
#pragma unroll
      for (int kf = 0; kf < 2; ++kf)
        pa[kf] = load_a_frag(&Ps[wave][l16][kf * 32], lh);
      // O += P * V ; B[k=kv][n=d] = Vt[d][kv]
#pragma unroll
      for (int t = 0; t < 8; ++t)
#pragma unroll
        for (int kf = 0; kf < 2; ++kf) {
          v16h vf = *(const v16h*)&Vt[t * 16 + l16][kf * 32 + lh * 16];
          o[t] = wmma_f16(pa[kf], vf, o[t]);
        }
    }
    __syncthreads();
  }

  float inv[8];
#pragma unroll
  for (int e = 0; e < 8; ++e) inv[e] = 1.0f / l[e];
#pragma unroll
  for (int t = 0; t < 8; ++t)
#pragma unroll
    for (int e = 0; e < 8; ++e) {
      int row = b * SEQ + qr0 + e + lh * 8;
      int col = h * HD + t * 16 + l16;
      ao[(size_t)row * DMODEL + col] = (_Float16)(o[t][e] * inv[e]);
    }
}

// ---------------------------------------------------------------- launch
extern "C" void kernel_launch(void* const* d_in, const int* in_sizes, int n_in,
                              void* d_out, int out_size, void* d_ws, size_t ws_size,
                              hipStream_t stream) {
  const float* x    = (const float*)d_in[0];
  const float* fcos = (const float*)d_in[1];
  const float* fsin = (const float*)d_in[2];
  const float* wq   = (const float*)d_in[3];
  const float* wk   = (const float*)d_in[4];
  const float* wv   = (const float*)d_in[5];
  const float* wo   = (const float*)d_in[6];
  float* out = (float*)d_out;

  _Float16* ws = (_Float16*)d_ws;
  size_t o = 0;
  _Float16* x_h  = ws + o; o += (size_t)MTOT * DMODEL;     // 16 MB
  _Float16* wq_h = ws + o; o += (size_t)DMODEL * DMODEL;   //  8 MB
  _Float16* wk_h = ws + o; o += (size_t)DMODEL * KVDIM;    //  2 MB
  _Float16* wv_h = ws + o; o += (size_t)DMODEL * KVDIM;    //  2 MB
  _Float16* wo_h = ws + o; o += (size_t)DMODEL * DMODEL;   //  8 MB
  _Float16* q_h  = ws + o; o += (size_t)MTOT * DMODEL;     // 16 MB
  _Float16* k_h  = ws + o; o += (size_t)MTOT * KVDIM;      //  4 MB
  _Float16* v_h  = ws + o; o += (size_t)MTOT * KVDIM;      //  4 MB
  _Float16* ao_h = ws + o;                                 // 16 MB

  auto cvt = [&](const float* s, _Float16* d, size_t n) {
    cvt_f32_to_f16<<<dim3((unsigned)(n / 1024)), 256, 0, stream>>>(s, d, n);
  };
  cvt(x,  x_h,  (size_t)MTOT * DMODEL);
  cvt(wq, wq_h, (size_t)DMODEL * DMODEL);
  cvt(wk, wk_h, (size_t)DMODEL * KVDIM);
  cvt(wv, wv_h, (size_t)DMODEL * KVDIM);
  cvt(wo, wo_h, (size_t)DMODEL * DMODEL);

  // projections
  gemm_f16_kernel<false><<<dim3(DMODEL / 128, MTOT / 128), 256, 0, stream>>>(x_h, wq_h, q_h, DMODEL, DMODEL);
  gemm_f16_kernel<false><<<dim3(KVDIM / 128, MTOT / 128), 256, 0, stream>>>(x_h, wk_h, k_h, DMODEL, KVDIM);
  gemm_f16_kernel<false><<<dim3(KVDIM / 128, MTOT / 128), 256, 0, stream>>>(x_h, wv_h, v_h, DMODEL, KVDIM);

  // RoPE on q (16 heads) and k (4 heads)
  rope_kernel<<<dim3((unsigned)((size_t)MTOT * NH  * (HD / 2) / 256)), 256, 0, stream>>>(q_h, fcos, fsin, NH);
  rope_kernel<<<dim3((unsigned)((size_t)MTOT * NKV * (HD / 2) / 256)), 256, 0, stream>>>(k_h, fcos, fsin, NKV);

  // causal flash attention
  flash_kernel<<<dim3(SEQ / 128, NH, BQ), 256, 0, stream>>>(q_h, k_h, v_h, ao_h);

  // output projection -> f32
  gemm_f16_kernel<true><<<dim3(DMODEL / 128, MTOT / 128), 256, 0, stream>>>(ao_h, wo_h, out, DMODEL, DMODEL);
}
